// SpatialAttention_23922967838794
// MI455X (gfx1250) — matrix operand order repaired and checked by reading, not compiled, not run on hardware
//
#include <hip/hip_runtime.h>

// Problem constants (match reference)
#define BB 8
#define CC 128
#define SS 128
#define EE 128
#define HH 8
#define PP 16

typedef __attribute__((ext_vector_type(16))) __bf16 v16bf;
typedef __attribute__((ext_vector_type(8)))  float  v8f;
typedef __attribute__((ext_vector_type(4)))  int    v4i;
typedef unsigned int       u32;
typedef unsigned short     u16;
typedef unsigned long long u64;
typedef __attribute__((ext_vector_type(4), aligned(16))) u32 u32x4;

union FragU { u32x4 q[2]; v16bf v; };

// ---- native bf16 conversion (backend selects hardware cvt) ------------------
__device__ __forceinline__ u16 f2bf(float x) {
  union { __bf16 h; u16 s; } u; u.h = (__bf16)x; return u.s;
}
__device__ __forceinline__ u32 pack2(float a, float b) {
  union { __bf16 h[2]; u32 w; } u;
  u.h[0] = (__bf16)a; u.h[1] = (__bf16)b; return u.w;
}

// ---- async global->LDS copy (CDNA5 GLOBAL_LOAD_ASYNC_TO_LDS_B128) ----------
#if defined(__AMDGCN__) && __has_builtin(__builtin_amdgcn_global_load_async_to_lds_b128) && \
    __has_builtin(__builtin_amdgcn_s_wait_asynccnt)
#define HAVE_ASYNC 1
#else
#define HAVE_ASYNC 0
#endif

#if HAVE_ASYNC
__device__ __forceinline__ __attribute__((address_space(3))) v4i* as_lds_v4(void* p) {
  // low 32 bits of a flat LDS address are the wave-relative LDS byte offset
  return (__attribute__((address_space(3))) v4i*)(u32)(u64)p;
}
__device__ __forceinline__ __attribute__((address_space(1))) v4i* as_glob_v4(const void* p) {
  return (__attribute__((address_space(1))) v4i*)(u64)p;
}
#endif

__device__ __forceinline__ void copy16_g2l(u32* lds, const u32x4* g) {
#if HAVE_ASYNC
  __builtin_amdgcn_global_load_async_to_lds_b128(as_glob_v4((const void*)g),
                                                 as_lds_v4((void*)lds), 0, 0);
#else
  *(u32x4*)lds = *g;
#endif
}
__device__ __forceinline__ void async_join() {
#if HAVE_ASYNC
  __builtin_amdgcn_s_wait_asynccnt(0);
#endif
}

// ---- WMMA fragment loads from LDS (16B-aligned -> ds_load_b128) ------------
// A operand (16x32 bf16), LDS row-major [m][k]; lane = M.
__device__ __forceinline__ v16bf load_fragA(const u32* base, int row0, int rs, int kk, int lane) {
  const u32* p = base + (row0 + (lane & 15)) * rs + kk * 16 + ((lane >> 4) << 2);
  FragU f;
  f.q[0] = *(const u32x4*)p;        // words 0..3
  f.q[1] = *(const u32x4*)(p + 8);  // words 8..11
  return f.v;
}
// B operand (32x16 bf16) from W^T stored [n][k]; lane = N; words 0..3 and 8..11 (same k order as A).
__device__ __forceinline__ v16bf load_fragB(const u32* base, int row0, int rs, int kk, int lane) {
  const u32* p = base + (row0 + (lane & 15)) * rs + kk * 16 + ((lane >> 4) << 2);
  FragU f;
  f.q[0] = *(const u32x4*)p;
  f.q[1] = *(const u32x4*)(p + 8);
  return f.v;
}

__device__ __forceinline__ v8f wmma_bf16(v16bf a, v16bf b, v8f c) {
  return __builtin_amdgcn_wmma_f32_16x16x32_bf16(false, a, false, b, (short)0, c, false, false);
}

// ---------------------------------------------------------------------------
// Kernel 1: per (b,c) compute Q = X@WqT, K = X@WkT, V = X@WvT  (128x128x128)
// Q,K -> [B,S,H,C,P] bf16 ; V -> [B,S,H,P,C] bf16 (pre-transposed B operand)
// ---------------------------------------------------------------------------
__global__ __launch_bounds__(256)
void qkv_kernel(const float* __restrict__ x,
                const float* __restrict__ Wq,
                const float* __restrict__ Wk,
                const float* __restrict__ Wv,
                u16* __restrict__ Qws, u16* __restrict__ Kws,
                u16* __restrict__ Vtws)
{
  extern __shared__ __align__(16) u32 sm[];
  u32* Xs  = sm;            // [s][e]  128x64 words (bf16 pairs)
  u32* Wqs = sm + 8192;     // [n][e]
  u32* Wks = sm + 16384;    // [n][e]
  u32* Wvs = sm + 24576;    // [n][e]   total 32768 words = 128 KB

  const int bc = blockIdx.x;
  const int b = bc >> 7, c = bc & 127;
  const int tid = threadIdx.x;
  const int lane = tid & 31, wv = tid >> 5;

  const float* xb = x + (size_t)bc * (SS * EE);
  __builtin_prefetch(xb + tid * 64, 0, 1);               // global_prefetch_b8
  __builtin_prefetch(Wk + (size_t)(tid & 63) * 256, 0, 1);

  for (int w = tid; w < 8192; w += 256) {
    int row = w >> 6, e0 = (w & 63) << 1;
    Xs[w] = pack2(xb[row * EE + e0], xb[row * EE + e0 + 1]);
    int h = row >> 4, p = row & 15;                      // n = h*16 + p
    const float* wq = Wq + (size_t)((h * CC + c) * PP + p) * EE;   // [h][c][p][e]
    Wqs[w] = pack2(wq[e0], wq[e0 + 1]);
    const float* wk = Wk + (size_t)h * EE * PP + p;                // [h][e][p]
    Wks[w] = pack2(wk[e0 * PP], wk[(e0 + 1) * PP]);
    const float* wvv = Wv + (size_t)h * EE * PP + p;
    Wvs[w] = pack2(wvv[e0 * PP], wvv[(e0 + 1) * PP]);
  }
  __syncthreads();

  const int mt = wv, ln = lane & 15, hi = (lane >> 4) & 1;
  v16bf a0 = load_fragA(Xs, mt * 16, 64, 0, lane);
  v16bf a1 = load_fragA(Xs, mt * 16, 64, 1, lane);
  v16bf a2 = load_fragA(Xs, mt * 16, 64, 2, lane);
  v16bf a3 = load_fragA(Xs, mt * 16, 64, 3, lane);

  const int s0 = mt * 16 + hi * 8;                        // M row for r=0

#pragma unroll
  for (int targ = 0; targ < 3; ++targ) {
    const u32* Ws = (targ == 0) ? Wqs : (targ == 1 ? Wks : Wvs);
    for (int nt = 0; nt < 8; ++nt) {                      // nt == head index
      v8f acc = {};
      acc = wmma_bf16(a0, load_fragB(Ws, nt * 16, 64, 0, lane), acc);
      acc = wmma_bf16(a1, load_fragB(Ws, nt * 16, 64, 1, lane), acc);
      acc = wmma_bf16(a2, load_fragB(Ws, nt * 16, 64, 2, lane), acc);
      acc = wmma_bf16(a3, load_fragB(Ws, nt * 16, 64, 3, lane), acc);
      if (targ == 2) {
        u16* vb = Vtws + (size_t)(((b * SS + s0) * HH + nt) * PP + ln) * CC + c;
#pragma unroll
        for (int r = 0; r < 8; ++r)
          vb[(size_t)r * (HH * PP * CC)] = f2bf(acc[r]); // imm-offset stores
      } else {
        u16* qb = ((targ == 0) ? Qws : Kws) +
                  (size_t)(((b * SS + s0) * HH + nt) * CC + c) * PP + ln;
#pragma unroll
        for (int r = 0; r < 8; ++r)
          qb[(size_t)r * (HH * CC * PP)] = f2bf(acc[r]);
      }
    }
  }
}

// ---------------------------------------------------------------------------
// Kernel 2: per (b,s,h): scores = Q Kᵀ / 4 ; softmax over joints ; O = A V
// ---------------------------------------------------------------------------
__global__ __launch_bounds__(256)
void attn_kernel(const u16* __restrict__ Qws, const u16* __restrict__ Kws,
                 const u16* __restrict__ Vtws, u16* __restrict__ Ows)
{
  extern __shared__ __align__(16) u32 sm[];
  u32*   Qp = sm;                       // 128 x 16 words, K padded to 32
  u32*   Kp = sm + 2048;                // 128 x 16 words
  float* Sc = (float*)(sm + 4096);      // 128 x 128 f32
  u32*   At = sm + 20480;               // 128 x 64 words (attn bf16)
  u32*   Vl = sm + 28672;               // 16  x 64 words (V^T bf16)

  const int blk = blockIdx.x;
  const int b = blk / (SS * HH);
  const int rem = blk % (SS * HH);
  const int s = rem / HH, h = rem % HH;
  const int tid = threadIdx.x, lane = tid & 31, wv = tid >> 5;

  const size_t base = (size_t)((b * SS + s) * HH + h);
  const u32x4* qsrc = (const u32x4*)((const u32*)Qws  + base * (CC * PP / 2));
  const u32x4* ksrc = (const u32x4*)((const u32*)Kws  + base * (CC * PP / 2));
  const u32x4* vsrc = (const u32x4*)((const u32*)Vtws + base * (PP * CC / 2));

  // async DMA the bf16 tiles into LDS; zero the K-padding with plain DS stores
  {
    const int row = tid >> 1, half = tid & 1;             // 256 threads = 128x2
    copy16_g2l(Qp + row * 16 + half * 4, qsrc + row * 2 + half);
    copy16_g2l(Kp + row * 16 + half * 4, ksrc + row * 2 + half);
    const u32x4 z = {0u, 0u, 0u, 0u};
    *(u32x4*)(Qp + row * 16 + 8 + half * 4) = z;
    *(u32x4*)(Kp + row * 16 + 8 + half * 4) = z;
    copy16_g2l(Vl + tid * 4, vsrc + tid);
  }
  async_join();
  __syncthreads();

  const int mt = wv, ln = lane & 15, hi = (lane >> 4) & 1;
  v16bf aq = load_fragA(Qp, mt * 16, 16, 0, lane);
  for (int nt = 0; nt < 8; ++nt) {
    v8f acc = {};
    acc = wmma_bf16(aq, load_fragB(Kp, nt * 16, 16, 0, lane), acc);
    float* sb = Sc + (mt * 16 + hi * 8) * CC + nt * 16 + ln;
#pragma unroll
    for (int r = 0; r < 8; ++r) sb[r * CC] = acc[r] * 0.25f;   // 1/sqrt(P)
  }
  __syncthreads();

  if (tid < CC) {                        // row-wise softmax over joints d
    float* row = Sc + tid * CC;
    float m = row[0];
    for (int d = 1; d < CC; ++d) m = fmaxf(m, row[d]);
    float sum = 0.f;
    for (int d = 0; d < CC; ++d) { float e = __expf(row[d] - m); row[d] = e; sum += e; }
    float inv = 1.f / sum;
    for (int j = 0; j < CC / 2; ++j)
      At[tid * 64 + j] = pack2(row[2 * j] * inv, row[2 * j + 1] * inv);
  }
  __syncthreads();

  // O = attn(128x128) @ V (K=d, N=p=16): one 16x16 tile per wave
  v8f acc = {};
#pragma unroll
  for (int kk = 0; kk < 4; ++kk) {
    v16bf aa = load_fragA(At, wv * 16, 64, kk, lane);
    v16bf bb = load_fragB(Vl, 0, 64, kk, lane);
    acc = wmma_bf16(aa, bb, acc);
  }
  const int c0 = wv * 16 + hi * 8;
  u16* ob = Ows + ((size_t)(b * CC + c0) * SS + s) * EE + h * PP + ln;
#pragma unroll
  for (int r = 0; r < 8; ++r)
    ob[(size_t)r * (SS * EE)] = f2bf(acc[r]);               // imm-offset stores
}

// ---------------------------------------------------------------------------
// Kernel 3: per (b,c): out = o'_bc (128x128) @ W_output[c]  (f32 result)
// ---------------------------------------------------------------------------
__global__ __launch_bounds__(256)
void out_kernel(const u16* __restrict__ Ows, const float* __restrict__ Wo,
                float* __restrict__ out)
{
  extern __shared__ __align__(16) u32 sm[];
  u32* Os  = sm;           // [s][e]  128x64 words
  u32* Wos = sm + 8192;    // [f][e]  (W_output transposed)

  const int bc = blockIdx.x;
  const int c = bc & 127;
  const int tid = threadIdx.x, lane = tid & 31, wv = tid >> 5;

  const u32x4* osrc = (const u32x4*)((const u32*)Ows + (size_t)bc * (SS * EE / 2));
  const float* wo = Wo + (size_t)c * EE * EE;              // [e][f]
  __builtin_prefetch(wo + tid * 64, 0, 1);                 // global_prefetch_b8

#pragma unroll
  for (int i = 0; i < 8; ++i)                              // async copy o' tile
    copy16_g2l(Os + (i * 256 + tid) * 4, osrc + i * 256 + tid);

  for (int w = tid; w < 8192; w += 256) {                  // transpose+convert W
    int f = w >> 6, e0 = (w & 63) << 1;
    Wos[w] = pack2(wo[e0 * EE + f], wo[(e0 + 1) * EE + f]);
  }
  async_join();
  __syncthreads();

  const int mt = wv, ln = lane & 15, hi = (lane >> 4) & 1;
  v16bf a0 = load_fragA(Os, mt * 16, 64, 0, lane);
  v16bf a1 = load_fragA(Os, mt * 16, 64, 1, lane);
  v16bf a2 = load_fragA(Os, mt * 16, 64, 2, lane);
  v16bf a3 = load_fragA(Os, mt * 16, 64, 3, lane);

  float* ob = out + (size_t)bc * (SS * EE);
  for (int nt = 0; nt < 8; ++nt) {
    v8f acc = {};
    acc = wmma_bf16(a0, load_fragB(Wos, nt * 16, 64, 0, lane), acc);
    acc = wmma_bf16(a1, load_fragB(Wos, nt * 16, 64, 1, lane), acc);
    acc = wmma_bf16(a2, load_fragB(Wos, nt * 16, 64, 2, lane), acc);
    acc = wmma_bf16(a3, load_fragB(Wos, nt * 16, 64, 3, lane), acc);
    float* obp = ob + (size_t)(mt * 16 + hi * 8) * EE + nt * 16 + ln;
#pragma unroll
    for (int r = 0; r < 8; ++r)
      obp[(size_t)r * EE] = acc[r];                        // imm-offset stores
  }
}

// ---------------------------------------------------------------------------
extern "C" void kernel_launch(void* const* d_in, const int* in_sizes, int n_in,
                              void* d_out, int out_size, void* d_ws, size_t ws_size,
                              hipStream_t stream) {
  (void)in_sizes; (void)n_in; (void)out_size; (void)ws_size;
  const float* x  = (const float*)d_in[0];
  const float* Wq = (const float*)d_in[1];
  const float* Wk = (const float*)d_in[2];
  const float* Wv = (const float*)d_in[3];
  const float* Wo = (const float*)d_in[4];
  float* out = (float*)d_out;

  // bf16 workspace: Q,K [B,S,H,C,P]; Vt [B,S,H,P,C]; o' [B,C,S,E]  (4 x 32 MiB)
  const size_t N = (size_t)BB * SS * HH * CC * PP;   // 16,777,216 elems
  u16* Qws  = (u16*)d_ws;
  u16* Kws  = Qws + N;
  u16* Vtws = Kws + N;
  u16* Ows  = Vtws + N;

  qkv_kernel <<<dim3(BB * CC),      256, 131072, stream>>>(x, Wq, Wk, Wv, Qws, Kws, Vtws);
  attn_kernel<<<dim3(BB * SS * HH), 256, 118784, stream>>>(Qws, Kws, Vtws, Ows);
  out_kernel <<<dim3(BB * CC),      256,  65536, stream>>>(Ows, Wo, out);
}
